// MiniMHSA_89893665505600
// MI455X (gfx1250) — compile-verified, matrix operand order checked by reading, not run on hardware
//
#include <hip/hip_runtime.h>
#include <hip/hip_bf16.h>

// CDNA5 / gfx1250, wave32. bf16 WMMA; flash attention with pre-transposed V.

typedef __attribute__((ext_vector_type(16))) __bf16 v16bf;
typedef __attribute__((ext_vector_type(8)))  __bf16 v8bf;
typedef __attribute__((ext_vector_type(8)))  float  v8f;

#define N_B 4
#define L_S 2048
#define D_M 1024
#define H_N 16
#define HD_ 64
#define E3  3072   // 3*D

static __device__ __forceinline__ v8f wmma_bf16(v16bf a, v16bf b, v8f c) {
  // D = A(16x32 bf16) * B(32x16 bf16) + C(16x16 f32)
  return __builtin_amdgcn_wmma_f32_16x16x32_bf16(
      /*neg_a=*/false, a, /*neg_b=*/false, b,
      /*c_mod=*/(short)0, c, /*reuse_a=*/false, /*reuse_b=*/false);
}

// ---------------------------------------------------------------------------
// f32 -> bf16 conversion (vectorized, 4 elems/thread)
// ---------------------------------------------------------------------------
__global__ __launch_bounds__(256) void cvt_f32_bf16(const float* __restrict__ in,
                                                    __bf16* __restrict__ out, int n4) {
  int i = blockIdx.x * blockDim.x + threadIdx.x;
  if (i < n4) {
    float4 f = ((const float4*)in)[i];
    out[4 * i + 0] = (__bf16)f.x;
    out[4 * i + 1] = (__bf16)f.y;
    out[4 * i + 2] = (__bf16)f.z;
    out[4 * i + 3] = (__bf16)f.w;
  }
}

// ---------------------------------------------------------------------------
// QKV projection GEMM (NT): qkv[m,e] = x[m,:] . Wqkv[e,:] + b[e]
// Q,K written row-major into qk [N*L, 2048] (bf16).
// V written TRANSPOSED into vt [N, 1024(head*dim), L] (bf16) so the flash
// kernel's P.V B-fragments become contiguous 32B loads.  The C-layout makes
// the transposed store 8 contiguous bf16 per lane (one 16B store).
// Each wave computes a 16x64 tile; 8 waves / block.
// ---------------------------------------------------------------------------
__global__ __launch_bounds__(256) void gemm_qkv(const __bf16* __restrict__ A,
                                                const __bf16* __restrict__ Bw,
                                                const float* __restrict__ bias,
                                                __bf16* __restrict__ qk,
                                                __bf16* __restrict__ vt) {
  const int K = 1024, Nc = 3072;
  const int lane = threadIdx.x & 31;
  const int wave = threadIdx.x >> 5;
  const int r16  = lane & 15;
  const int half = lane >> 4;
  const int tilesN = Nc >> 6;                 // 48
  const int gw = blockIdx.x * 8 + wave;
  const int tm = (gw / tilesN) << 4;          // global row tile (n*2048 + l)
  const int tn = (gw % tilesN) << 6;

  v8f acc[4] = {};
  const __bf16* aRow = A + (size_t)(tm + r16) * K + half * 16;
  const __bf16* bRow = Bw + (size_t)(tn + r16) * K + half * 16;
  for (int kk = 0; kk < K; kk += 32) {
    v16bf aF = *(const v16bf*)(aRow + kk);
#pragma unroll
    for (int t = 0; t < 4; ++t) {
      v16bf bF = *(const v16bf*)(bRow + (size_t)t * 16 * K + kk);
      acc[t] = wmma_bf16(aF, bF, acc[t]);
    }
  }

  const int n    = tm >> 11;        // batch index
  const int lloc = (tm & 2047) + 8 * half;
#pragma unroll
  for (int t = 0; t < 4; ++t) {
    const int col = tn + t * 16 + r16;
    const float bv = bias[col];
    if (tn + t * 16 < 2048) {
      // Q or K slab: row-major [N*L, 2048]
#pragma unroll
      for (int rr = 0; rr < 8; ++rr)
        qk[(size_t)(tm + rr + 8 * half) * 2048 + col] = (__bf16)(acc[t][rr] + bv);
    } else {
      // V slab: transposed [n][h*64+d][l]; 8 contiguous bf16 per lane
      const int hd = col - 2048;
      v8bf pack;
#pragma unroll
      for (int rr = 0; rr < 8; ++rr) pack[rr] = (__bf16)(acc[t][rr] + bv);
      *(v8bf*)(vt + ((size_t)n * 1024 + hd) * L_S + lloc) = pack;
    }
  }
}

// ---------------------------------------------------------------------------
// Output projection GEMM (NT): out[m,e] = O[m,:] . Wout[e,:] + b[e], f32 out.
// ---------------------------------------------------------------------------
__global__ __launch_bounds__(256) void gemm_out(const __bf16* __restrict__ A,
                                                const __bf16* __restrict__ Bw,
                                                const float* __restrict__ bias,
                                                float* __restrict__ Cout) {
  const int K = 1024, Nc = 1024;
  const int lane = threadIdx.x & 31;
  const int wave = threadIdx.x >> 5;
  const int r16  = lane & 15;
  const int half = lane >> 4;
  const int tilesN = Nc >> 6;                 // 16
  const int gw = blockIdx.x * 8 + wave;
  const int tm = (gw / tilesN) << 4;
  const int tn = (gw % tilesN) << 6;

  v8f acc[4] = {};
  const __bf16* aRow = A + (size_t)(tm + r16) * K + half * 16;
  const __bf16* bRow = Bw + (size_t)(tn + r16) * K + half * 16;
  for (int kk = 0; kk < K; kk += 32) {
    v16bf aF = *(const v16bf*)(aRow + kk);
#pragma unroll
    for (int t = 0; t < 4; ++t) {
      v16bf bF = *(const v16bf*)(bRow + (size_t)t * 16 * K + kk);
      acc[t] = wmma_bf16(aF, bF, acc[t]);
    }
  }
#pragma unroll
  for (int t = 0; t < 4; ++t) {
    const int col = tn + t * 16 + r16;
    const float bv = bias[col];
#pragma unroll
    for (int rr = 0; rr < 8; ++rr)
      Cout[(size_t)(tm + rr + 8 * half) * Nc + col] = acc[t][rr] + bv;
  }
}

// ---------------------------------------------------------------------------
// Flash attention. Each wave owns a 16-query tile of one (n,h); streams 32-key
// tiles. QK^T via WMMA (contiguous A/B frags from row-major Q/K); online
// softmax in registers; P round-trips through 1KB of per-wave LDS to convert
// C-layout -> A-layout; P.V B-frags load directly from pre-transposed V.
// global_prefetch on next tile's K/V lines.  8 waves / block.
// ---------------------------------------------------------------------------
__global__ __launch_bounds__(256) void flash_attn(const __bf16* __restrict__ qk,
                                                  const __bf16* __restrict__ vt,
                                                  const unsigned char* __restrict__ mask,
                                                  __bf16* __restrict__ Obuf) {
  __shared__ __bf16 Plds[8][16 * 32];   // 1 KB / wave

  const int lane = threadIdx.x & 31;
  const int wave = threadIdx.x >> 5;
  const int r16  = lane & 15;
  const int half = lane >> 4;
  const int h = blockIdx.y;
  const int n = blockIdx.z;
  const int qBase = (blockIdx.x * 8 + wave) * 16;

  const __bf16* qbase = qk + (size_t)n * L_S * 2048;                 // rows = l
  const __bf16* vbase = vt + ((size_t)n * 1024 + h * HD_) * L_S;     // [d][l]
  const unsigned char* mrow = mask + (size_t)n * L_S;

  // Preload Q A-fragments (loop invariant): depth 64 = 2 chunks of 32
  v16bf aq[2];
#pragma unroll
  for (int dc = 0; dc < 2; ++dc)
    aq[dc] = *(const v16bf*)(qbase + (size_t)(qBase + r16) * 2048 +
                             h * HD_ + dc * 32 + half * 16);

  v8f o[4] = {};
  float m[8], l[8];
#pragma unroll
  for (int rr = 0; rr < 8; ++rr) { m[rr] = -1e30f; l[rr] = 0.0f; }

  for (int kt = 0; kt < L_S / 32; ++kt) {
    const int kBase = kt * 32;

    // Prefetch next tile's K rows and V lines into cache
    if (kt + 1 < L_S / 32) {
      __builtin_prefetch(qbase + (size_t)(kBase + 32 + lane) * 2048 + 1024 + h * HD_, 0, 3);
      __builtin_prefetch(vbase + (size_t)(lane & 63) * L_S + kBase + 32, 0, 3);
    }

    v8f s[2];
#pragma unroll
    for (int c = 0; c < 2; ++c) {
      v8f acc = {};
#pragma unroll
      for (int dc = 0; dc < 2; ++dc) {
        // B[kdepth, keycol]: lane = key column, 16 contiguous depth elems
        v16bf bk = *(const v16bf*)(qbase + (size_t)(kBase + 16 * c + r16) * 2048 +
                                   1024 + h * HD_ + dc * 32 + half * 16);
        acc = wmma_bf16(aq[dc], bk, acc);
      }
      const bool mk = mrow[kBase + 16 * c + r16] != 0;   // column mask
#pragma unroll
      for (int rr = 0; rr < 8; ++rr)
        acc[rr] = mk ? -1e9f : acc[rr] * 0.125f;         // scale = 1/sqrt(64)
      s[c] = acc;
    }

    // Online softmax: row stats across the 16-lane column group
    float corr[8];
#pragma unroll
    for (int rr = 0; rr < 8; ++rr) {
      float rm = fmaxf(s[0][rr], s[1][rr]);
      rm = fmaxf(rm, __shfl_xor(rm, 1, 32));
      rm = fmaxf(rm, __shfl_xor(rm, 2, 32));
      rm = fmaxf(rm, __shfl_xor(rm, 4, 32));
      rm = fmaxf(rm, __shfl_xor(rm, 8, 32));
      const float mn = fmaxf(m[rr], rm);
      const float cr = __expf(m[rr] - mn);
      const float p0 = __expf(s[0][rr] - mn);
      const float p1 = __expf(s[1][rr] - mn);
      s[0][rr] = p0; s[1][rr] = p1;
      float rs = p0 + p1;
      rs += __shfl_xor(rs, 1, 32);
      rs += __shfl_xor(rs, 2, 32);
      rs += __shfl_xor(rs, 4, 32);
      rs += __shfl_xor(rs, 8, 32);
      l[rr] = l[rr] * cr + rs;
      m[rr] = mn;
      corr[rr] = cr;
    }
#pragma unroll
    for (int t = 0; t < 4; ++t)
#pragma unroll
      for (int rr = 0; rr < 8; ++rr) o[t][rr] *= corr[rr];

    // Stage P (16x32) in LDS: C-layout store, A-layout read
    __bf16* P = Plds[wave];
#pragma unroll
    for (int c = 0; c < 2; ++c)
#pragma unroll
      for (int rr = 0; rr < 8; ++rr)
        P[(rr + 8 * half) * 32 + 16 * c + r16] = (__bf16)s[c][rr];

    v16bf aP = *(const v16bf*)(P + r16 * 32 + half * 16);

    // O += P(16x32) * V(32x64): B-frags straight from transposed V in global
#pragma unroll
    for (int t = 0; t < 4; ++t) {
      v16bf bV = *(const v16bf*)(vbase + (size_t)(t * 16 + r16) * L_S +
                                 kBase + half * 16);
      o[t] = wmma_bf16(aP, bV, o[t]);
    }
  }

  // Epilogue: normalize and store bf16 to [N*L, D] at this head's slot
#pragma unroll
  for (int rr = 0; rr < 8; ++rr) l[rr] = 1.0f / l[rr];
#pragma unroll
  for (int t = 0; t < 4; ++t)
#pragma unroll
    for (int rr = 0; rr < 8; ++rr) {
      const size_t row = (size_t)n * L_S + qBase + rr + 8 * half;
      Obuf[row * D_M + h * HD_ + t * 16 + r16] = (__bf16)(o[t][rr] * l[rr]);
    }
}

// ---------------------------------------------------------------------------
extern "C" void kernel_launch(void* const* d_in, const int* in_sizes, int n_in,
                              void* d_out, int out_size, void* d_ws, size_t ws_size,
                              hipStream_t stream) {
  (void)in_sizes; (void)n_in; (void)out_size; (void)ws_size;

  const float* x            = (const float*)d_in[0];
  const unsigned char* mask = (const unsigned char*)d_in[1];  // jnp bool = 1 byte
  const float* W_qkv        = (const float*)d_in[2];
  const float* b_qkv        = (const float*)d_in[3];
  const float* W_out        = (const float*)d_in[4];
  const float* b_out        = (const float*)d_in[5];

  // Workspace layout (~92 MB total)
  char* ws = (char*)d_ws;
  __bf16* xbf  = (__bf16*)(ws);                          // 16.8 MB
  __bf16* wqbf = (__bf16*)(ws + 16777216);               //  6.3 MB
  __bf16* wobf = (__bf16*)(ws + 23068672);               //  2.1 MB
  __bf16* qkbf = (__bf16*)(ws + 25165824);               // 33.6 MB  [N*L, 2048]
  __bf16* vtbf = (__bf16*)(ws + 58720256);               // 16.8 MB  [N,1024,L]
  __bf16* obf  = (__bf16*)(ws + 75497472);               // 16.8 MB  [N*L, 1024]

  // 1) f32 -> bf16 conversions
  cvt_f32_bf16<<<8388608 / 4 / 256, 256, 0, stream>>>(x, xbf, 8388608 / 4);
  cvt_f32_bf16<<<3145728 / 4 / 256, 256, 0, stream>>>(W_qkv, wqbf, 3145728 / 4);
  cvt_f32_bf16<<<1048576 / 4 / 256, 256, 0, stream>>>(W_out, wobf, 1048576 / 4);

  // 2) QKV projection -> qk row-major + V transposed
  gemm_qkv<<<(8192 / 16) * (3072 / 64) / 8, 256, 0, stream>>>(
      xbf, wqbf, b_qkv, qkbf, vtbf);

  // 3) Flash attention: grid = (L/128 qtile-groups, H, N), 8 waves/block
  dim3 g(L_S / 128, H_N, N_B);
  flash_attn<<<g, 256, 0, stream>>>(qkbf, vtbf, mask, obf);

  // 4) Output projection -> f32 d_out
  gemm_out<<<(8192 / 16) * (1024 / 64) / 8, 256, 0, stream>>>(
      obf, wobf, b_out, (float*)d_out);
}
